// LSTM_torchscript_38293928411400
// MI455X (gfx1250) — matrix operand order repaired and checked by reading, not compiled, not run on hardware
//
#include <hip/hip_runtime.h>

// ---------------------------------------------------------------------------
// MI455X (gfx1250, wave32) fused 2-layer LSTM.
//  - per-wave batch tile of 16 rows, h/c state resident across T=60 steps
//  - recurrent matmuls via v_wmma_f32_16x16x32_f16 (f16 in, f32 accum)
//  - rnn1out handed to kernel2 through d_ws, pre-packed in WMMA A-operand
//    order (f16) so kernel2 loads fragments with 4 coalesced b128 loads
//  - out head folded: out = h2 @ (Wout@Wlat)^T + (Wout@blat + bout)
// ---------------------------------------------------------------------------

typedef __attribute__((ext_vector_type(16))) _Float16 v16h;
typedef __attribute__((ext_vector_type(8)))  float    v8f;

union AF { v16h h; uint4 q[2]; };

__device__ __forceinline__ float sigm(float x)      { return 1.0f / (1.0f + __expf(-x)); }
__device__ __forceinline__ float tanh_fast(float x) { return 2.0f / (1.0f + __expf(-2.0f * x)) - 1.0f; }

__device__ __forceinline__ v8f wmma16(const AF& a, const AF& b, v8f c) {
  return __builtin_amdgcn_wmma_f32_16x16x32_f16(false, a.h, false, b.h, (short)0, c, false, false);
}

// Problem constants
#define NB   16384
#define NT   60
#define NH   64
#define NG   256   // 4*H

// Workspace layout (bytes)
#define WS_WHH1P  ((size_t)0)        // 16384 f16 = 32768 B  (B-operand packed)
#define WS_WIH2P  ((size_t)32768)    // 32768 B
#define WS_WHH2P  ((size_t)65536)    // 32768 B
#define WS_WF     ((size_t)98304)    // Wout@Wlat, 256 f32 = 1024 B
#define WS_BF     ((size_t)99328)    // 4 f32
#define WS_RNN1   ((size_t)131072)   // 1024 tiles * 60 steps * 2048 B = 125829120 B

// Packed B-operand layout convention (self-consistent producer/consumer):
//   fragment q = (j*4 + gate)*2 + kh, j in [0,4) column-group, gate in {i,f,g,o},
//   kh = K-half. Within a fragment (512 f16): lane l holds 16 contiguous f16 at
//   l*16;  semantic: n = gate*64 + j*16 + (l&15), k = kh*32 + (jj>>3)*16 + (l>>4)*8 + (jj&7).
// A-operand (16x32 f16): lane l row M=l&15; v16h = K[kh*32 + (l>>4? 8:0) + 0..7]
//   ++ K[kh*32 + 16 + (l>>4? 8:0) + 0..7]   (two 16B chunks from row-major LDS).
// C/D layout: vgpr r, lane l -> M = r + 8*(l>>4), N = (group)*16 + (l&15).

__global__ void k_pack(const float* __restrict__ Whh1, const float* __restrict__ Wih2,
                       const float* __restrict__ Whh2, const float* __restrict__ Wlat,
                       const float* __restrict__ blat, const float* __restrict__ Wout,
                       const float* __restrict__ bout, char* __restrict__ ws) {
  _Float16* whh1p = (_Float16*)(ws + WS_WHH1P);
  _Float16* wih2p = (_Float16*)(ws + WS_WIH2P);
  _Float16* whh2p = (_Float16*)(ws + WS_WHH2P);
  float*    wf    = (float*)(ws + WS_WF);
  float*    bf    = (float*)(ws + WS_BF);
  const int t = threadIdx.x;  // 256 threads, 1 block
  for (int p = t; p < 16384; p += 256) {
    const int q  = p >> 9, w = p & 511;
    const int l  = w >> 4, jj = w & 15;
    const int jt = q >> 3, gate = (q >> 1) & 3, kh = q & 1;
    const int n  = gate * 64 + jt * 16 + (l & 15);
    const int k  = kh * 32 + (jj >> 3) * 16 + (l >> 4) * 8 + (jj & 7);
    whh1p[p] = (_Float16)Whh1[n * NH + k];
    wih2p[p] = (_Float16)Wih2[n * NH + k];
    whh2p[p] = (_Float16)Whh2[n * NH + k];
  }
  if (t < 256) {  // Wf = Wout @ Wlat  (4x64)
    const int ny = t >> 6, k = t & 63;
    float s = 0.f;
    for (int j = 0; j < 64; ++j) s += Wout[ny * 64 + j] * Wlat[j * 64 + k];
    wf[t] = s;
  }
  if (t < 4) {
    float s = bout[t];
    for (int j = 0; j < 64; ++j) s += Wout[t * 64 + j] * blat[j];
    bf[t] = s;
  }
}

// ---------------------------------------------------------------------------
// Kernel 1: LSTM1 top-down (reads inputs_main flipped in time), writes h1 to
// ws in A-packed f16 order at original time index (which is LSTM2's order).
// ---------------------------------------------------------------------------
__global__ __launch_bounds__(128) void k_lstm1(
    const float* __restrict__ xmain, const float* __restrict__ aux,
    const float* __restrict__ Ws1, const float* __restrict__ bs1,
    const float* __restrict__ Ws2, const float* __restrict__ bs2,
    const float* __restrict__ Wih1, const float* __restrict__ bih1,
    const float* __restrict__ bhh1, char* __restrict__ ws) {
  __shared__ __align__(16) _Float16 sW[16384];      // Whh1 packed (B-operand)
  __shared__ __align__(16) float4   sWx[256];       // Wih1 rows as float4
  __shared__ float  sGb[256];                       // bih1 + bhh1
  __shared__ float  sWs1[192], sBs1[64], sWs2[192], sBs2[64];
  __shared__ __align__(16) float4   sXa[4][16];     // aux per wave-tile
  __shared__ __align__(16) float4   sX[4][16];      // x_t per wave-tile
  __shared__ __align__(16) _Float16 sStage[4][1024];// h staging (16x64 f16)

  const int tid = threadIdx.x;
  {
    const uint4* src = (const uint4*)(ws + WS_WHH1P);
    uint4* dst = (uint4*)sW;
    for (int p = tid; p < 2048; p += 128) dst[p] = src[p];
  }
  for (int p = tid; p < 256; p += 128) {
    sWx[p] = *(const float4*)&Wih1[p * 4];
    sGb[p] = bih1[p] + bhh1[p];
  }
  for (int p = tid; p < 192; p += 128) { sWs1[p] = Ws1[p]; sWs2[p] = Ws2[p]; }
  if (tid < 64) { sBs1[tid] = bs1[tid]; sBs2[tid] = bs2[tid]; }

  const int wave = tid >> 5, l = tid & 31, lm = l & 15, lh = l >> 4;
  const int tile = blockIdx.x * 4 + wave;
  const int rowbase = tile * 16;
  if (lh == 0) {
    const float* ar = aux + (size_t)(rowbase + lm) * 3;
    sXa[wave][lm] = make_float4(ar[0], ar[1], ar[2], 0.f);
  }
  __syncthreads();

  // initial h = aux@Ws1^T + bs1  -> staged f16 [m][k]
  for (int idx = l; idx < 1024; idx += 32) {
    const int m = idx >> 6, k = idx & 63;
    const float4 xa = sXa[wave][m];
    const float v = sBs1[k] + xa.x * sWs1[k * 3] + xa.y * sWs1[k * 3 + 1] + xa.z * sWs1[k * 3 + 2];
    sStage[wave][idx] = (_Float16)v;
  }
  // initial c = tanh(aux@Ws2^T + bs2) in C-fragment layout
  v8f cfr[4];
#pragma unroll
  for (int r = 0; r < 8; ++r) {
    const float4 xa = sXa[wave][r + 8 * lh];
#pragma unroll
    for (int j = 0; j < 4; ++j) {
      const int n = j * 16 + lm;
      const float v = sBs2[n] + xa.x * sWs2[n * 3] + xa.y * sWs2[n * 3 + 1] + xa.z * sWs2[n * 3 + 2];
      cfr[j][r] = tanh_fast(v);
    }
  }
  asm volatile("s_wait_dscnt 0x0" ::: "memory");
  AF alo, ahi;
  alo.q[0] = *(const uint4*)&sStage[wave][lm * 64 + lh * 8];
  alo.q[1] = *(const uint4*)&sStage[wave][lm * 64 + lh * 8 + 16];
  ahi.q[0] = *(const uint4*)&sStage[wave][lm * 64 + 32 + lh * 8];
  ahi.q[1] = *(const uint4*)&sStage[wave][lm * 64 + 32 + lh * 8 + 16];

  char* const slotbase = ws + WS_RNN1 + (size_t)tile * NT * 2048;

  for (int s = 0; s < NT; ++s) {
    const int torig = NT - 1 - s;  // LSTM1 runs over flipped time
    if (lh == 0)
      sX[wave][lm] = *(const float4*)&xmain[((size_t)(rowbase + lm) * NT + torig) * 4];
    asm volatile("s_wait_dscnt 0x0" ::: "memory");
    float4 xr[8];
#pragma unroll
    for (int r = 0; r < 8; ++r) xr[r] = sX[wave][r + 8 * lh];

#pragma unroll
    for (int j = 0; j < 4; ++j) {
      v8f g4[4];
#pragma unroll
      for (int gt = 0; gt < 4; ++gt) {
        const int n = gt * 64 + j * 16 + lm;
        const float b = sGb[n];
        const float4 wv = sWx[n];
        v8f acc;
#pragma unroll
        for (int r = 0; r < 8; ++r)
          acc[r] = b + wv.x * xr[r].x + wv.y * xr[r].y + wv.z * xr[r].z + wv.w * xr[r].w;
        const int fb = (j * 4 + gt) * 1024;
        AF blo, bhi;
        blo.q[0] = *(const uint4*)&sW[fb + l * 16];
        blo.q[1] = *(const uint4*)&sW[fb + l * 16 + 8];
        bhi.q[0] = *(const uint4*)&sW[fb + 512 + l * 16];
        bhi.q[1] = *(const uint4*)&sW[fb + 512 + l * 16 + 8];
        acc = wmma16(alo, blo, acc);
        acc = wmma16(ahi, bhi, acc);
        g4[gt] = acc;
      }
#pragma unroll
      for (int r = 0; r < 8; ++r) {
        const float iv = sigm(g4[0][r]), fv = sigm(g4[1][r]);
        const float gv = tanh_fast(g4[2][r]), ov = sigm(g4[3][r]);
        const float cv = fv * cfr[j][r] + iv * gv;
        cfr[j][r] = cv;
        const float hv = ov * tanh_fast(cv);
        sStage[wave][(r + 8 * lh) * 64 + j * 16 + lm] = (_Float16)hv;
      }
    }
    asm volatile("s_wait_dscnt 0x0" ::: "memory");
    alo.q[0] = *(const uint4*)&sStage[wave][lm * 64 + lh * 8];
    alo.q[1] = *(const uint4*)&sStage[wave][lm * 64 + lh * 8 + 16];
    ahi.q[0] = *(const uint4*)&sStage[wave][lm * 64 + 32 + lh * 8];
    ahi.q[1] = *(const uint4*)&sStage[wave][lm * 64 + 32 + lh * 8 + 16];
    // store h1 (A-packed, f16) to ws at original time index, coalesced
    char* slot = slotbase + (size_t)torig * 2048;
    *(uint4*)(slot + l * 32)          = alo.q[0];
    *(uint4*)(slot + l * 32 + 16)     = alo.q[1];
    *(uint4*)(slot + 1024 + l * 32)   = ahi.q[0];
    *(uint4*)(slot + 1024 + l * 32 + 16) = ahi.q[1];
  }
}

// ---------------------------------------------------------------------------
// Kernel 2: LSTM2 bottom-up + fused output heads.
// Whh2 packed in LDS (latency-critical recurrent path); Wih2 packed streamed
// from L2 (32KB hot) to stay under the static-LDS limit.
// ---------------------------------------------------------------------------
__global__ __launch_bounds__(128) void k_lstm2(
    const float* __restrict__ hx2, const float* __restrict__ cx2,
    const float* __restrict__ bih2, const float* __restrict__ bhh2,
    const float* __restrict__ Wsfc, const float* __restrict__ bsfc,
    char* __restrict__ ws, float* __restrict__ out) {
  __shared__ __align__(16) _Float16 sWhh[16384];    // Whh2 packed
  __shared__ float sGb[256], sWf[256], sBf[4], sWsfc[192], sBsfc[4];
  __shared__ __align__(16) _Float16 sStage[4][1024];

  const int tid = threadIdx.x;
  {
    const uint4* src = (const uint4*)(ws + WS_WHH2P);
    uint4* dst = (uint4*)sWhh;
    for (int p = tid; p < 2048; p += 128) dst[p] = src[p];
  }
  for (int p = tid; p < 256; p += 128) {
    sGb[p] = bih2[p] + bhh2[p];
    sWf[p] = ((const float*)(ws + WS_WF))[p];
  }
  for (int p = tid; p < 192; p += 128) sWsfc[p] = Wsfc[p];
  if (tid < 4) { sBf[tid] = ((const float*)(ws + WS_BF))[tid]; sBsfc[tid] = (tid < 3) ? bsfc[tid] : 0.f; }

  const int wave = tid >> 5, l = tid & 31, lm = l & 15, lh = l >> 4;
  const int tile = blockIdx.x * 4 + wave;
  const int rowbase = tile * 16;

  // initial h2 staging (coalesced) and c2 fragments
  for (int idx = l; idx < 1024; idx += 32)
    sStage[wave][idx] = (_Float16)hx2[(size_t)rowbase * 64 + idx];
  v8f cfr[4];
#pragma unroll
  for (int r = 0; r < 8; ++r)
#pragma unroll
    for (int j = 0; j < 4; ++j)
      cfr[j][r] = cx2[(size_t)rowbase * 64 + (r + 8 * lh) * 64 + j * 16 + lm];
  __syncthreads();

  AF alo, ahi;
  alo.q[0] = *(const uint4*)&sStage[wave][lm * 64 + lh * 8];
  alo.q[1] = *(const uint4*)&sStage[wave][lm * 64 + lh * 8 + 16];
  ahi.q[0] = *(const uint4*)&sStage[wave][lm * 64 + 32 + lh * 8];
  ahi.q[1] = *(const uint4*)&sStage[wave][lm * 64 + 32 + lh * 8 + 16];

  const char* const slotbase = ws + WS_RNN1 + (size_t)tile * NT * 2048;
  const _Float16* const wih = (const _Float16*)(ws + WS_WIH2P);

  // software-pipelined x fragments
  AF xlo, xhi;
  {
    const char* slot = slotbase;
    xlo.q[0] = *(const uint4*)(slot + l * 32);
    xlo.q[1] = *(const uint4*)(slot + l * 32 + 16);
    xhi.q[0] = *(const uint4*)(slot + 1024 + l * 32);
    xhi.q[1] = *(const uint4*)(slot + 1024 + l * 32 + 16);
  }

  for (int t = 0; t < NT; ++t) {
#pragma unroll
    for (int j = 0; j < 4; ++j) {
      v8f g4[4];
#pragma unroll
      for (int gt = 0; gt < 4; ++gt) {
        const int n = gt * 64 + j * 16 + lm;
        const float b = sGb[n];
        v8f acc = {b, b, b, b, b, b, b, b};
        const int fb = (j * 4 + gt) * 1024;
        AF bxl, bxh, bhl, bhh;
        bxl.q[0] = *(const uint4*)&wih[fb + l * 16];
        bxl.q[1] = *(const uint4*)&wih[fb + l * 16 + 8];
        bxh.q[0] = *(const uint4*)&wih[fb + 512 + l * 16];
        bxh.q[1] = *(const uint4*)&wih[fb + 512 + l * 16 + 8];
        bhl.q[0] = *(const uint4*)&sWhh[fb + l * 16];
        bhl.q[1] = *(const uint4*)&sWhh[fb + l * 16 + 8];
        bhh.q[0] = *(const uint4*)&sWhh[fb + 512 + l * 16];
        bhh.q[1] = *(const uint4*)&sWhh[fb + 512 + l * 16 + 8];
        acc = wmma16(alo, bhl, acc);
        acc = wmma16(ahi, bhh, acc);
        acc = wmma16(xlo, bxl, acc);
        acc = wmma16(xhi, bxh, acc);
        g4[gt] = acc;
      }
#pragma unroll
      for (int r = 0; r < 8; ++r) {
        const float iv = sigm(g4[0][r]), fv = sigm(g4[1][r]);
        const float gv = tanh_fast(g4[2][r]), ov = sigm(g4[3][r]);
        const float cv = fv * cfr[j][r] + iv * gv;
        cfr[j][r] = cv;
        const float hv = ov * tanh_fast(cv);
        sStage[wave][(r + 8 * lh) * 64 + j * 16 + lm] = (_Float16)hv;
      }
    }
    asm volatile("s_wait_dscnt 0x0" ::: "memory");
    alo.q[0] = *(const uint4*)&sStage[wave][lm * 64 + lh * 8];
    alo.q[1] = *(const uint4*)&sStage[wave][lm * 64 + lh * 8 + 16];
    ahi.q[0] = *(const uint4*)&sStage[wave][lm * 64 + 32 + lh * 8];
    ahi.q[1] = *(const uint4*)&sStage[wave][lm * 64 + 32 + lh * 8 + 16];
    // prefetch next step's x fragments
    {
      const int tn = (t < NT - 1) ? t + 1 : t;
      const char* slot = slotbase + (size_t)tn * 2048;
      xlo.q[0] = *(const uint4*)(slot + l * 32);
      xlo.q[1] = *(const uint4*)(slot + l * 32 + 16);
      xhi.q[0] = *(const uint4*)(slot + 1024 + l * 32);
      xhi.q[1] = *(const uint4*)(slot + 1024 + l * 32 + 16);
    }
    // fused head: out[m][ny] = h2[m] . Wf[ny] + bf[ny]   (2 outputs per lane)
    {
      const int m = l >> 1, ny0 = (l & 1) * 2;
      float o0 = sBf[ny0], o1 = sBf[ny0 + 1];
#pragma unroll
      for (int kk = 0; kk < 8; ++kk) {
        union { uint4 q; _Float16 e[8]; } u;
        u.q = *(const uint4*)&sStage[wave][m * 64 + kk * 8];
#pragma unroll
        for (int e = 0; e < 8; ++e) {
          const float hf = (float)u.e[e];
          o0 += hf * sWf[ny0 * 64 + kk * 8 + e];
          o1 += hf * sWf[ny0 * 64 + 64 + kk * 8 + e];
        }
      }
      *(float2*)&out[((size_t)(rowbase + m) * NT + t) * 4 + ny0] = make_float2(o0, o1);
    }
  }
  // out_sfc = relu(last_h @ Wsfc^T + bsfc)
  float* const osfc = out + (size_t)NB * NT * 4;
  for (int idx = l; idx < 48; idx += 32) {
    const int m = idx / 3, sc = idx - m * 3;
    float a = sBsfc[sc];
    for (int k = 0; k < 64; ++k)
      a += (float)sStage[wave][m * 64 + k] * sWsfc[sc * 64 + k];
    osfc[(size_t)(rowbase + m) * 3 + sc] = fmaxf(a, 0.f);
  }
}

extern "C" void kernel_launch(void* const* d_in, const int* in_sizes, int n_in,
                              void* d_out, int out_size, void* d_ws, size_t ws_size,
                              hipStream_t stream) {
  (void)in_sizes; (void)n_in; (void)out_size; (void)ws_size;
  char* ws = (char*)d_ws;
  k_pack<<<1, 256, 0, stream>>>(
      (const float*)d_in[9], (const float*)d_in[12], (const float*)d_in[13],
      (const float*)d_in[16], (const float*)d_in[17], (const float*)d_in[18],
      (const float*)d_in[19], ws);
  k_lstm1<<<NB / 64, 128, 0, stream>>>(
      (const float*)d_in[0], (const float*)d_in[1],
      (const float*)d_in[4], (const float*)d_in[5],
      (const float*)d_in[6], (const float*)d_in[7],
      (const float*)d_in[8], (const float*)d_in[10], (const float*)d_in[11], ws);
  k_lstm2<<<NB / 64, 128, 0, stream>>>(
      (const float*)d_in[2], (const float*)d_in[3],
      (const float*)d_in[14], (const float*)d_in[15],
      (const float*)d_in[20], (const float*)d_in[21],
      ws, (float*)d_out);
}
